// MultiheadAttention_3779571220983
// MI455X (gfx1250) — matrix operand order repaired and checked by reading, not compiled, not run on hardware
//
#include <hip/hip_runtime.h>

// ---------------------------------------------------------------------------
// MultiheadAttention with pre-softmax residual bias, CDNA5 (gfx1250) WMMA.
// T=4096, B=1, E=256, H=8, D=32.
//
// Roofline: residual_attn stream (8*4096*4096*4 B = 537 MB, read once) is the
// bound (~23 us @ 23.3 TB/s). All matmuls run as v_wmma_f32_16x16x32_bf16.
// Residual loads are non-temporal so the reused K/V/Q/Ao working set stays in
// the 192 MB L2 while the one-shot bias streams through.
// ---------------------------------------------------------------------------

typedef __attribute__((ext_vector_type(16))) __bf16 v16bf;
typedef __attribute__((ext_vector_type(8)))  __bf16 v8bf;
typedef __attribute__((ext_vector_type(8)))  float  v8f;

#define T_SEQ 4096
#define NH    8
#define HD    32
#define EMB   256
#define LDS_STRIDE 40   // halfs per row; 80B pitch -> bank-conflict-free, 16B aligned

static __device__ __forceinline__ v16bf cat8(v8bf lo, v8bf hi) {
  return __builtin_shufflevector(lo, hi, 0,1,2,3,4,5,6,7,8,9,10,11,12,13,14,15);
}

// convert two fp32 octets into one bf16 A/B fragment
static __device__ __forceinline__ v16bf cvt16(v8f a, v8f b) {
  v16bf r;
#pragma unroll
  for (int i = 0; i < 8; ++i) { r[i] = (__bf16)a[i]; r[8 + i] = (__bf16)b[i]; }
  return r;
}

static __device__ __forceinline__ v8f wmma_bf16(v16bf a, v16bf b, v8f c) {
  return __builtin_amdgcn_wmma_f32_16x16x32_bf16(
      /*neg_a=*/false, a, /*neg_b=*/false, b,
      /*c_mod=*/(short)0, c, /*reuse_a=*/false, /*reuse_b=*/false);
}

static __device__ __forceinline__ float rowmax16(float x) {
  x = fmaxf(x, __shfl_xor(x, 1, 32));
  x = fmaxf(x, __shfl_xor(x, 2, 32));
  x = fmaxf(x, __shfl_xor(x, 4, 32));
  x = fmaxf(x, __shfl_xor(x, 8, 32));
  return x;
}
static __device__ __forceinline__ float rowsum16(float x) {
  x += __shfl_xor(x, 1, 32);
  x += __shfl_xor(x, 2, 32);
  x += __shfl_xor(x, 4, 32);
  x += __shfl_xor(x, 8, 32);
  return x;
}

// ---------------------------------------------------------------------------
// Kernel 1: fused Q/K/V projection.  One wave = one 16x16 output tile of one
// of the three GEMMs [4096x256] @ [256x256]^T, K=256 in 8 bf16 WMMAs.
//   Q -> Qh[h][t][d] (bf16, pre-scaled by d^-0.5)
//   K -> Kh[h][t][d] (bf16)
//   V -> Vt[h*32+d][t] (bf16, transposed for the P@V B-operand)
// ---------------------------------------------------------------------------
__global__ __launch_bounds__(256)
void qkv_proj_kernel(const float* __restrict__ query,
                     const float* __restrict__ value,
                     const float* __restrict__ Wq, const float* __restrict__ bq,
                     const float* __restrict__ Wk, const float* __restrict__ bk,
                     const float* __restrict__ Wv, const float* __restrict__ bv,
                     __bf16* __restrict__ Qh, __bf16* __restrict__ Kh,
                     __bf16* __restrict__ Vt)
{
  const int lane = threadIdx.x & 31;
  const int wave = threadIdx.x >> 5;
  int task = blockIdx.x * 8 + wave;        // 3 * 4096 tasks
  const int mat = task >> 12;              // 0=Q 1=K 2=V
  task &= 4095;
  const int t0 = (task >> 4) << 4;
  const int e0 = (task & 15) << 4;
  const int n  = lane & 15;
  const int hi = lane >> 4;
  const int b0 = hi ? 8 : 0;

  const float* X    = (mat == 2) ? value : query;          // key arg is ignored
  const float* W    = (mat == 0) ? Wq : (mat == 1) ? Wk : Wv;
  const float* bias = (mat == 0) ? bq : (mat == 1) ? bk : bv;

  // A operand: row m = t0+n, contiguous K chunks at b0 and b0+16
  const float* xrow = X + (size_t)(t0 + n) * EMB;
  // B operand: column e = e0+n, K chunks at b0 and b0+16 (same as A)
  const float* wrow = W + (size_t)(e0 + n) * EMB + (hi ? 8 : 0);

  v8f acc = {};
#pragma unroll
  for (int k = 0; k < EMB; k += 32) {
    v16bf a = cvt16(*(const v8f*)(xrow + k + b0),
                    *(const v8f*)(xrow + k + b0 + 16));
    v16bf b = cvt16(*(const v8f*)(wrow + k),
                    *(const v8f*)(wrow + k + 16));
    acc = wmma_bf16(a, b, acc);
  }

  const int   e    = e0 + n;
  const float bval = bias[e];
  const float scal = 0.17677669529663687f;   // 32^-0.5

  if (mat == 0) {
    const int h = e >> 5, d = e & 31;
#pragma unroll
    for (int v = 0; v < 8; ++v) {
      const int t = t0 + v + 8 * hi;
      Qh[((size_t)(h * T_SEQ + t)) * HD + d] = (__bf16)((acc[v] + bval) * scal);
    }
  } else if (mat == 1) {
    const int h = e >> 5, d = e & 31;
#pragma unroll
    for (int v = 0; v < 8; ++v) {
      const int t = t0 + v + 8 * hi;
      Kh[((size_t)(h * T_SEQ + t)) * HD + d] = (__bf16)(acc[v] + bval);
    }
  } else {
#pragma unroll
    for (int v = 0; v < 8; ++v) {
      const int t = t0 + v + 8 * hi;
      Vt[(size_t)e * T_SEQ + t] = (__bf16)(acc[v] + bval);
    }
  }
}

// ---------------------------------------------------------------------------
// Kernel 2: flash attention with residual bias.  One wave = 16 query rows of
// one head, full D=32 accumulator, streaming over S in 32-wide tiles.
//   scores  : v_wmma_f32_16x16x32_bf16 (Q x K^T) + fp32 residual (NT loads)
//   softmax : online, 16-lane shfl-xor row reductions
//   P @ V   : P staged bf16 through padded LDS (C-layout -> A-layout)
// ---------------------------------------------------------------------------
__global__ __launch_bounds__(256)
void flash_attn_kernel(const __bf16* __restrict__ Qh,
                       const __bf16* __restrict__ Kh,
                       const __bf16* __restrict__ Vt,
                       const float*  __restrict__ res,
                       __bf16* __restrict__ Ao)
{
  __shared__ __align__(16) __bf16 pl[8][16 * LDS_STRIDE];

  const int lane = threadIdx.x & 31;
  const int wave = threadIdx.x >> 5;
  const int task = blockIdx.x * 8 + wave;     // 8 heads * 256 q-tiles = 2048
  const int h  = task >> 8;
  const int q0 = (task & 255) << 4;
  const int n  = lane & 15;
  const int hi = lane >> 4;
  const int b0 = hi ? 8 : 0;

  // Q fragment (A operand), held in registers for the whole sweep
  const __bf16* qrow = Qh + ((size_t)(h * T_SEQ + q0 + n)) * HD;
  const v16bf qa = cat8(*(const v8bf*)(qrow + 2 * b0),
                        *(const v8bf*)(qrow + 2 * b0 + 8));

  v8f o0 = {}, o1 = {};                       // O tile, d in [0,16) and [16,32)
  float mrun[8], lrun[8];
#pragma unroll
  for (int v = 0; v < 8; ++v) { mrun[v] = -1.0e30f; lrun[v] = 0.0f; }

  const float*  resbase = res + ((size_t)h * T_SEQ + q0) * T_SEQ
                              + (size_t)(hi * 8) * T_SEQ + n;
  __bf16* const myp = &pl[wave][0];
  const __bf16* vcol = Vt + ((size_t)(h * HD + n)) * T_SEQ + (hi ? 16 : 0);

  for (int s0 = 0; s0 < T_SEQ; s0 += 32) {
    // ---- K B-operands: rows s0+n and s0+16+n, contiguous d (hi half +16)
    const __bf16* kp = Kh + ((size_t)(h * T_SEQ + s0 + n)) * HD + (hi ? 16 : 0);
    const v16bf kb0 = *(const v16bf*)(kp);
    const v16bf kb1 = *(const v16bf*)(kp + 16 * HD);

    v8f z = {};
    v8f sA = wmma_bf16(qa, kb0, z);
    v8f sB = wmma_bf16(qa, kb1, z);

    // ---- residual tile in exact C-matrix lane layout.  This is the dominant
    // HBM stream (537 MB, read-once): non-temporal so it bypasses L2 retention
    // and leaves the reused K/V/P/O working set cached.
    float r0[8], r1[8];
    const float* rp = resbase + s0;
#pragma unroll
    for (int v = 0; v < 8; ++v) {
      r0[v] = __builtin_nontemporal_load(rp + (size_t)v * T_SEQ);
      r1[v] = __builtin_nontemporal_load(rp + (size_t)v * T_SEQ + 16);
    }

    // ---- V B-operands (contiguous in s thanks to transposed Vt)
    const v16bf vb0 = *(const v16bf*)(vcol + s0);
    const v16bf vb1 = *(const v16bf*)(vcol + (size_t)16 * T_SEQ + s0);

    // ---- online softmax, per owned row (m = v + 8*hi)
#pragma unroll
    for (int v = 0; v < 8; ++v) {
      const float t0 = sA[v] + r0[v];
      const float t1 = sB[v] + r1[v];
      const float mnew  = fmaxf(mrun[v], rowmax16(fmaxf(t0, t1)));
      const float p0    = __expf(t0 - mnew);
      const float p1    = __expf(t1 - mnew);
      const float rs    = rowsum16(p0 + p1);
      const float alpha = __expf(mrun[v] - mnew);
      mrun[v] = mnew;
      lrun[v] = lrun[v] * alpha + rs;
      o0[v] *= alpha;  o1[v] *= alpha;
      const int row = v + 8 * hi;
      myp[row * LDS_STRIDE + n]      = (__bf16)p0;
      myp[row * LDS_STRIDE + 16 + n] = (__bf16)p1;
    }

    // ---- P: C-layout -> A-layout via LDS (same wave; DScnt handled by compiler)
    const __bf16* prow = myp + n * LDS_STRIDE;
    const v16bf pa = cat8(*(const v8bf*)(prow + 2 * b0),
                          *(const v8bf*)(prow + 2 * b0 + 8));
    o0 = wmma_bf16(pa, vb0, o0);
    o1 = wmma_bf16(pa, vb1, o1);
  }

  // ---- normalize and emit attention output [t][e] (bf16, A-friendly layout)
#pragma unroll
  for (int v = 0; v < 8; ++v) {
    const int   m   = v + 8 * hi;
    const float inv = 1.0f / lrun[v];
    __bf16* dst = Ao + (size_t)(q0 + m) * EMB + h * HD + n;
    dst[0]  = (__bf16)(o0[v] * inv);
    dst[16] = (__bf16)(o1[v] * inv);
  }
}

// ---------------------------------------------------------------------------
// Kernel 3: output projection  out = Ao @ Wo^T + bo   (fp32 result, NT store)
// ---------------------------------------------------------------------------
__global__ __launch_bounds__(256)
void out_proj_kernel(const __bf16* __restrict__ Ao,
                     const float*  __restrict__ Wo,
                     const float*  __restrict__ bo,
                     float* __restrict__ out)
{
  const int lane = threadIdx.x & 31;
  const int wave = threadIdx.x >> 5;
  const int task = blockIdx.x * 8 + wave;     // 256 t-tiles * 16 e-tiles
  const int t0 = (task >> 4) << 4;
  const int e0 = (task & 15) << 4;
  const int n  = lane & 15;
  const int hi = lane >> 4;
  const int b0 = hi ? 8 : 0;

  const __bf16* arow = Ao + (size_t)(t0 + n) * EMB;
  const float*  wrow = Wo + (size_t)(e0 + n) * EMB + (hi ? 8 : 0);

  v8f acc = {};
#pragma unroll
  for (int k = 0; k < EMB; k += 32) {
    v16bf a = cat8(*(const v8bf*)(arow + k + b0),
                   *(const v8bf*)(arow + k + b0 + 16));
    v16bf b = cvt16(*(const v8f*)(wrow + k),
                    *(const v8f*)(wrow + k + 16));
    acc = wmma_bf16(a, b, acc);
  }

  const float bval = bo[e0 + n];
#pragma unroll
  for (int v = 0; v < 8; ++v) {
    // write-once result: non-temporal store, don't dirty L2
    __builtin_nontemporal_store(acc[v] + bval,
        out + (size_t)(t0 + v + 8 * hi) * EMB + e0 + n);
  }
}

// ---------------------------------------------------------------------------
extern "C" void kernel_launch(void* const* d_in, const int* in_sizes, int n_in,
                              void* d_out, int out_size, void* d_ws, size_t ws_size,
                              hipStream_t stream) {
  const float* query = (const float*)d_in[0];
  // d_in[1] ("key") is ignored by the reference forward (q and k from query)
  const float* value = (const float*)d_in[2];
  const float* res   = (const float*)d_in[3];
  const float* Wq    = (const float*)d_in[4];
  const float* bq    = (const float*)d_in[5];
  const float* Wk    = (const float*)d_in[6];
  const float* bk    = (const float*)d_in[7];
  const float* Wv    = (const float*)d_in[8];
  const float* bv    = (const float*)d_in[9];
  const float* Wo    = (const float*)d_in[10];
  const float* bo    = (const float*)d_in[11];
  float* out = (float*)d_out;

  char* ws = (char*)d_ws;
  const size_t SZ = (size_t)NH * T_SEQ * HD * sizeof(__bf16);   // 2 MB each
  __bf16* Qh = (__bf16*)(ws);
  __bf16* Kh = (__bf16*)(ws + SZ);
  __bf16* Vt = (__bf16*)(ws + 2 * SZ);
  __bf16* Ao = (__bf16*)(ws + 3 * SZ);

  // 3 GEMMs * 4096 tiles / 8 waves per block
  qkv_proj_kernel<<<1536, 256, 0, stream>>>(query, value, Wq, bq, Wk, bk,
                                            Wv, bv, Qh, Kh, Vt);
  // 8 heads * 256 q-tiles / 8 waves per block
  flash_attn_kernel<<<256, 256, 0, stream>>>(Qh, Kh, Vt, res, Ao);
  // 4096 tiles / 8 waves per block
  out_proj_kernel<<<512, 256, 0, stream>>>(Ao, Wo, bo, out);
}
